// Decoder_86242943304025
// MI455X (gfx1250) — compile-verified, matrix operand order checked by reading, not run on hardware
//
#include <hip/hip_runtime.h>
#include <hip/hip_bf16.h>

// ---------------------------------------------------------------------------
// Pointer-network decoder for MI455X (gfx1250, wave32, WMMA).
// All GEMMs: v_wmma_f32_16x16x32_bf16 (bf16 operands, f32 accumulate).
// Projected context e_g/e_p kept in BF16 to halve the dominant HBM stream.
// ---------------------------------------------------------------------------

#define Bn   1024
#define Ln   256
#define En   256
#define Hn   256
#define H4n  1024

typedef __attribute__((ext_vector_type(16))) __bf16       v16bf;
typedef __attribute__((ext_vector_type(8)))  float        v8f;
typedef __attribute__((ext_vector_type(4)))  unsigned int uint4v;

// ---------------- bf16 helpers (RNE) ----------------
__device__ __forceinline__ __bf16 f32_to_bf16(float f) {
    union { float f; unsigned int u; } in;
    in.f = f;
    unsigned int u = in.u;
    u = (u + 0x7FFFu + ((u >> 16) & 1u)) >> 16;
    union { unsigned short s; __bf16 b; } out;
    out.s = (unsigned short)u;
    return out.b;
}
__device__ __forceinline__ float bf16_to_f32(__bf16 b) {
    union { unsigned short s; __bf16 b; } in;
    in.b = b;
    union { unsigned int u; float f; } out;
    out.u = ((unsigned int)in.s) << 16;
    return out.f;
}
__device__ __forceinline__ float sigmoidf(float x) { return 1.0f / (1.0f + expf(-x)); }

// hash -> uniform(0,1)
__device__ __forceinline__ float rnd01(unsigned int a, unsigned int b) {
    unsigned int x = a * 747796405u + b * 2891336453u + 1013904223u;
    x ^= x >> 16; x *= 0x7feb352du; x ^= x >> 15; x *= 0x846ca68bu; x ^= x >> 16;
    return (float)(x >> 8) * (1.0f / 16777216.0f) + 1.0f / 33554432.0f;
}

// load one 16x32 bf16 WMMA fragment row slice: two aligned 16B chunks
__device__ __forceinline__ v16bf load_frag(const __bf16* __restrict__ p) {
    union { uint4v u[2]; v16bf v; } f;
    f.u[0] = *(const uint4v*)(p);
    f.u[1] = *(const uint4v*)(p + 16);
    return f.v;
}

// ---------------- generic conversion / setup kernels ----------------
__global__ void k_f32_to_bf16(const float* __restrict__ in, __bf16* __restrict__ out, long n) {
    long i = (long)blockIdx.x * blockDim.x + threadIdx.x;
    if (i < n) out[i] = f32_to_bf16(in[i]);
}

// out[n*K + k] = bf16(in[k*N + n])   (weight [K,N] -> BT [N,K])
__global__ void k_transpose_cvt(const float* __restrict__ in, __bf16* __restrict__ out, int K, int N) {
    long t = (long)blockIdx.x * blockDim.x + threadIdx.x;
    if (t >= (long)K * N) return;
    int n = (int)(t % N);
    int k = (int)(t / N);
    out[(long)n * K + k] = f32_to_bf16(in[(long)k * N + n]);
}

__global__ void k_bias_sum(const float* __restrict__ a, const float* __restrict__ b,
                           float* __restrict__ o, int n) {
    int i = blockIdx.x * blockDim.x + threadIdx.x;
    if (i < n) o[i] = a[i] + b[i];
}

__global__ void k_init_state(const float* __restrict__ dec, const float* __restrict__ hx,
                             const float* __restrict__ cx,
                             __bf16* __restrict__ xb, __bf16* __restrict__ hb,
                             float* __restrict__ hf, float* __restrict__ cf) {
    int i = blockIdx.x * blockDim.x + threadIdx.x;  // B*H
    if (i >= Bn * Hn) return;
    xb[i] = f32_to_bf16(dec[i]);
    hb[i] = f32_to_bf16(hx[i]);
    hf[i] = hx[i];
    cf[i] = cx[i];
}

__global__ void k_init_mask(int* __restrict__ mask, int* __restrict__ prev) {
    long i = (long)blockIdx.x * blockDim.x + threadIdx.x;
    if (i < (long)Bn * Ln) mask[i] = 0;
    if (i < Bn) prev[i] = -1;
}

__global__ void k_update_mask(const int* __restrict__ prev, int* __restrict__ mask) {
    int b = blockIdx.x * blockDim.x + threadIdx.x;
    if (b >= Bn) return;
    int p = prev[b];
    if (p >= 0) mask[(long)b * Ln + p] = 1;
}

// ---------------- WMMA GEMM: C[M,N] = A[M,K](bf16,rm) * BT[N,K](bf16,rm) ----
// ACC=1: accumulate into existing f32 C (bias ignored). BOUT=1: bf16 output.
template <int ACC, int BOUT>
__global__ __launch_bounds__(256)
void k_gemm_wmma(const __bf16* __restrict__ A, const __bf16* __restrict__ BT,
                 const float* __restrict__ bias, void* __restrict__ Cout,
                 long M, int N, int K) {
    const int wave = threadIdx.x >> 5;
    const int lane = threadIdx.x & 31;
    const long tile = (long)blockIdx.x * 8 + wave;
    const int tilesN = N >> 4;
    const long tm = (tile / tilesN) << 4;
    const int  tn = (int)(tile % tilesN) << 4;
    if (tm >= M) return;                       // wave-uniform

    const int mn    = lane & 15;               // row of A frag / row of BT frag
    const int kb    = (lane & 16) ? 8 : 0;     // K sub-base per half-wave
    const int mbase = (lane & 16) ? 8 : 0;     // C rows per half-wave

    v8f c;
    if (ACC) {
        const float* C = (const float*)Cout;
#pragma unroll
        for (int r = 0; r < 8; ++r) c[r] = C[(tm + mbase + r) * (long)N + tn + mn];
    } else {
        float bv = bias ? bias[tn + mn] : 0.0f;
#pragma unroll
        for (int r = 0; r < 8; ++r) c[r] = bv;
    }

    const __bf16* arow = A  + (tm + mn) * (long)K + kb;
    const __bf16* brow = BT + (long)(tn + mn) * K + kb;
#pragma unroll 2
    for (int kq = 0; kq < K; kq += 32) {
        v16bf a = load_frag(arow + kq);
        v16bf b = load_frag(brow + kq);
        c = __builtin_amdgcn_wmma_f32_16x16x32_bf16(false, a, false, b,
                                                    (short)0, c, false, false);
    }

    if (BOUT) {
        __bf16* C = (__bf16*)Cout;
#pragma unroll
        for (int r = 0; r < 8; ++r)
            C[(tm + mbase + r) * (long)N + tn + mn] = f32_to_bf16(c[r]);
    } else {
        float* C = (float*)Cout;
#pragma unroll
        for (int r = 0; r < 8; ++r)
            C[(tm + mbase + r) * (long)N + tn + mn] = c[r];
    }
}

// ---------------- LSTM elementwise ----------------
__global__ void k_lstm(const float* __restrict__ gates, float* __restrict__ c,
                       float* __restrict__ hf, __bf16* __restrict__ hb) {
    int i = blockIdx.x * blockDim.x + threadIdx.x;  // B*H
    if (i >= Bn * Hn) return;
    int b = i >> 8, h = i & 255;
    long base = (long)b * H4n;
    float ig = sigmoidf(gates[base + h]);
    float fg = sigmoidf(gates[base + Hn + h]);
    float gg = tanhf(gates[base + 2 * Hn + h]);
    float og = sigmoidf(gates[base + 3 * Hn + h]);
    float cy = fg * c[i] + ig * gg;
    float hy = og * tanhf(cy);
    c[i] = cy;
    hf[i] = hy;
    hb[i] = f32_to_bf16(hy);
}

// ---------------- attention scores: u[b,l] = sum_k v[k]*tanh(q[b,k]+e[l,b,k])
__global__ __launch_bounds__(256)
void k_attn_scores(const float* __restrict__ q, const __bf16* __restrict__ e,
                   const float* __restrict__ v, float* __restrict__ u) {
    const int lane = threadIdx.x & 31;
    const long w = (long)blockIdx.x * 8 + (threadIdx.x >> 5);
    const int b = (int)(w >> 8);      // / Ln
    const int l = (int)(w & 255);     // % Ln
    const __bf16* erow = e + ((long)l * Bn + b) * Hn + lane * 8;
    const float*  qrow = q + (long)b * Hn + lane * 8;
    const float*  vrow = v + lane * 8;
    float s = 0.0f;
#pragma unroll
    for (int j = 0; j < 8; ++j)
        s += vrow[j] * tanhf(qrow[j] + bf16_to_f32(erow[j]));
#pragma unroll
    for (int off = 16; off > 0; off >>= 1)
        s += __shfl_xor(s, off, 32);
    if (lane == 0) u[(long)b * Ln + l] = s;
}

// ---------------- masked softmax over L (wave per row) ----------------
__global__ __launch_bounds__(256)
void k_masked_softmax(const float* __restrict__ u, const int* __restrict__ mask,
                      float* __restrict__ p) {
    const int lane = threadIdx.x & 31;
    const int b = blockIdx.x * 8 + (threadIdx.x >> 5);
    const long base = (long)b * Ln;
    float vals[8];
    float mx = -__builtin_inff();
#pragma unroll
    for (int j = 0; j < 8; ++j) {
        int l = j * 32 + lane;
        float x = u[base + l];
        if (mask[base + l]) x = -__builtin_inff();
        vals[j] = x;
        mx = fmaxf(mx, x);
    }
#pragma unroll
    for (int off = 16; off > 0; off >>= 1) mx = fmaxf(mx, __shfl_xor(mx, off, 32));
    float sum = 0.0f;
#pragma unroll
    for (int j = 0; j < 8; ++j) { vals[j] = expf(vals[j] - mx); sum += vals[j]; }
#pragma unroll
    for (int off = 16; off > 0; off >>= 1) sum += __shfl_xor(sum, off, 32);
    float inv = 1.0f / sum;
#pragma unroll
    for (int j = 0; j < 8; ++j) p[base + j * 32 + lane] = vals[j] * inv;
}

// ---------------- glimpse: g[b,k] = sum_l p[b,l] * e[l,b,k] ----------------
__global__ __launch_bounds__(256)
void k_glimpse(const float* __restrict__ p, const __bf16* __restrict__ e,
               float* __restrict__ gf, __bf16* __restrict__ gb) {
    int t = blockIdx.x * blockDim.x + threadIdx.x;  // B*H
    if (t >= Bn * Hn) return;
    int b = t >> 8, k = t & 255;
    const long pbase = (long)b * Ln;
    float acc = 0.0f;
    for (int l = 0; l < Ln; ++l) {
        if (l + 4 < Ln)
            __builtin_prefetch((const void*)(e + ((long)(l + 4) * Bn + b) * Hn + k), 0, 1);
        acc += p[pbase + l] * bf16_to_f32(e[((long)l * Bn + b) * Hn + k]);
    }
    gf[t] = acc;
    gb[t] = f32_to_bf16(acc);
}

// ---------------- pointer head: clip, mask, softmax, sample ----------------
__global__ __launch_bounds__(256)
void k_pointer_finalize(const float* __restrict__ up, const int* __restrict__ mask,
                        float* __restrict__ out_probs, int* __restrict__ out_sel,
                        int* __restrict__ prev, int t) {
    __shared__ float sm[256];
    const int l = threadIdx.x;
    const int b = blockIdx.x;
    const long base = (long)b * Ln;
    float logit = 10.0f * tanhf(up[base + l]);
    if (mask[base + l]) logit = -__builtin_inff();
    sm[l] = logit; __syncthreads();
    for (int s = 128; s > 0; s >>= 1) { if (l < s) sm[l] = fmaxf(sm[l], sm[l + s]); __syncthreads(); }
    float mx = sm[0]; __syncthreads();
    float ex = expf(logit - mx);
    sm[l] = ex; __syncthreads();
    for (int s = 128; s > 0; s >>= 1) { if (l < s) sm[l] += sm[l + s]; __syncthreads(); }
    float sum = sm[0]; __syncthreads();
    float prob = ex / sum;
    out_probs[((long)t * Bn + b) * Ln + l] = prob;
    sm[l] = prob; __syncthreads();
    if (l == 0) {
        float u = rnd01((unsigned int)t + 42u, (unsigned int)b);
        float acc = 0.0f; int idx = 0;
        for (int j = 0; j < Ln; ++j) {
            acc += sm[j];
            idx = j;
            if (u <= acc) break;
        }
        prev[b] = idx;
        out_sel[(long)t * Bn + b] = idx;
    }
}

// ---------------- next input gather ----------------
__global__ void k_gather_x(const int* __restrict__ prev, const float* __restrict__ emb,
                           __bf16* __restrict__ xb) {
    int t = blockIdx.x * blockDim.x + threadIdx.x;  // B*E
    if (t >= Bn * En) return;
    int b = t >> 8, e = t & 255;
    int idx = prev[b];
    xb[t] = f32_to_bf16(emb[((long)idx * Bn + b) * En + e]);
}

__global__ void k_final_copy(const float* __restrict__ hf, const float* __restrict__ cf,
                             float* __restrict__ out_hy, float* __restrict__ out_cy) {
    int i = blockIdx.x * blockDim.x + threadIdx.x;
    if (i >= Bn * Hn) return;
    out_hy[i] = hf[i];
    out_cy[i] = cf[i];
}

// ---------------------------------------------------------------------------
extern "C" void kernel_launch(void* const* d_in, const int* in_sizes, int n_in,
                              void* d_out, int out_size, void* d_ws, size_t ws_size,
                              hipStream_t stream) {
    (void)in_sizes; (void)n_in; (void)out_size; (void)ws_size;

    const float* dec   = (const float*)d_in[0];
    const float* emb   = (const float*)d_in[1];
    const float* hx    = (const float*)d_in[2];
    const float* cx    = (const float*)d_in[3];
    const float* ctx   = (const float*)d_in[4];
    const float* Wi    = (const float*)d_in[5];
    const float* bi    = (const float*)d_in[6];
    const float* Wh    = (const float*)d_in[7];
    const float* bh    = (const float*)d_in[8];
    const float* Wq_g  = (const float*)d_in[9];
    const float* bq_g  = (const float*)d_in[10];
    const float* Wr_g  = (const float*)d_in[11];
    const float* br_g  = (const float*)d_in[12];
    const float* v_g   = (const float*)d_in[13];
    const float* Wq_p  = (const float*)d_in[14];
    const float* bq_p  = (const float*)d_in[15];
    const float* Wr_p  = (const float*)d_in[16];
    const float* br_p  = (const float*)d_in[17];
    const float* v_p   = (const float*)d_in[18];

    // ---- workspace layout (bytes) ----
    char* W = (char*)d_ws;
    const size_t SZ_LBH_BF = (size_t)Ln * Bn * Hn * 2;   // 134217728
    size_t off = 0;
    __bf16* ctx_b = (__bf16*)(W + off); off += SZ_LBH_BF;
    __bf16* e_g   = (__bf16*)(W + off); off += SZ_LBH_BF;
    __bf16* e_p   = (__bf16*)(W + off); off += SZ_LBH_BF;
    __bf16* WiT   = (__bf16*)(W + off); off += (size_t)En * H4n * 2;
    __bf16* WhT   = (__bf16*)(W + off); off += (size_t)Hn * H4n * 2;
    __bf16* WqgT  = (__bf16*)(W + off); off += (size_t)Hn * Hn * 2;
    __bf16* WqpT  = (__bf16*)(W + off); off += (size_t)Hn * Hn * 2;
    __bf16* WrgT  = (__bf16*)(W + off); off += (size_t)Hn * Hn * 2;
    __bf16* WrpT  = (__bf16*)(W + off); off += (size_t)Hn * Hn * 2;
    __bf16* x_b   = (__bf16*)(W + off); off += (size_t)Bn * En * 2;
    __bf16* h_b   = (__bf16*)(W + off); off += (size_t)Bn * Hn * 2;
    float*  h_f   = (float*) (W + off); off += (size_t)Bn * Hn * 4;
    float*  c_f   = (float*) (W + off); off += (size_t)Bn * Hn * 4;
    float*  gates = (float*) (W + off); off += (size_t)Bn * H4n * 4;
    float*  qg    = (float*) (W + off); off += (size_t)Bn * Hn * 4;
    float*  qp    = (float*) (W + off); off += (size_t)Bn * Hn * 4;
    float*  gv_f  = (float*) (W + off); off += (size_t)Bn * Hn * 4;
    __bf16* gv_b  = (__bf16*)(W + off); off += (size_t)Bn * Hn * 2;
    float*  ug    = (float*) (W + off); off += (size_t)Bn * Ln * 4;
    float*  pg    = (float*) (W + off); off += (size_t)Bn * Ln * 4;
    float*  up    = (float*) (W + off); off += (size_t)Bn * Ln * 4;
    int*    mask  = (int*)   (W + off); off += (size_t)Bn * Ln * 4;
    int*    prev  = (int*)   (W + off); off += (size_t)Bn * 4;
    float*  bsum  = (float*) (W + off); off += (size_t)H4n * 4;

    // ---- output layout (elements): probs[L,B,L] f32 | sel[L,B] i32 | hy | cy
    float* out_probs = (float*)d_out;
    const long OFF_SEL = (long)Ln * Bn * Ln;                 // 67108864
    int*   out_sel   = (int*)d_out + OFF_SEL;
    float* out_hy    = (float*)d_out + OFF_SEL + (long)Ln * Bn;
    float* out_cy    = out_hy + (long)Bn * Hn;

    const int TB = 256;
    // ---- one-time setup ----
    {
        long n = (long)Ln * Bn * Hn;
        k_f32_to_bf16<<<(unsigned)((n + TB - 1) / TB), TB, 0, stream>>>(ctx, ctx_b, n);
    }
    k_transpose_cvt<<<(En * H4n + TB - 1) / TB, TB, 0, stream>>>(Wi,   WiT,  En, H4n);
    k_transpose_cvt<<<(Hn * H4n + TB - 1) / TB, TB, 0, stream>>>(Wh,   WhT,  Hn, H4n);
    k_transpose_cvt<<<(Hn * Hn  + TB - 1) / TB, TB, 0, stream>>>(Wq_g, WqgT, Hn, Hn);
    k_transpose_cvt<<<(Hn * Hn  + TB - 1) / TB, TB, 0, stream>>>(Wq_p, WqpT, Hn, Hn);
    k_transpose_cvt<<<(Hn * Hn  + TB - 1) / TB, TB, 0, stream>>>(Wr_g, WrgT, Hn, Hn);
    k_transpose_cvt<<<(Hn * Hn  + TB - 1) / TB, TB, 0, stream>>>(Wr_p, WrpT, Hn, Hn);
    k_bias_sum<<<(H4n + TB - 1) / TB, TB, 0, stream>>>(bi, bh, bsum, H4n);
    k_init_state<<<(Bn * Hn + TB - 1) / TB, TB, 0, stream>>>(dec, hx, cx, x_b, h_b, h_f, c_f);
    k_init_mask<<<(Bn * Ln + TB - 1) / TB, TB, 0, stream>>>(mask, prev);

    // e_g / e_p : [L*B, H] = ctx_b[L*B, H] x WrT[H, H] + br  (bf16 output)
    {
        long M = (long)Ln * Bn;
        unsigned grid = (unsigned)((M / 16) * (Hn / 16) / 8);
        k_gemm_wmma<0, 1><<<grid, TB, 0, stream>>>(ctx_b, WrgT, br_g, (void*)e_g, M, Hn, Hn);
        k_gemm_wmma<0, 1><<<grid, TB, 0, stream>>>(ctx_b, WrpT, br_p, (void*)e_p, M, Hn, Hn);
    }

    const unsigned g_gates = (unsigned)((Bn / 16) * (H4n / 16) / 8);  // 512
    const unsigned g_hh    = (unsigned)((Bn / 16) * (Hn / 16) / 8);   // 128
    const unsigned g_bh    = (Bn * Hn + TB - 1) / TB;                 // 1024
    const unsigned g_attn  = (unsigned)((long)Bn * Ln / 8);           // 32768 blocks of 8 waves

    // ---- 256 decode steps ----
    for (int t = 0; t < Ln; ++t) {
        // gates = x@Wi + (bi+bh); gates += h@Wh
        k_gemm_wmma<0, 0><<<g_gates, TB, 0, stream>>>(x_b, WiT, bsum, (void*)gates, Bn, H4n, En);
        k_gemm_wmma<1, 0><<<g_gates, TB, 0, stream>>>(h_b, WhT, nullptr, (void*)gates, Bn, H4n, Hn);
        k_lstm<<<g_bh, TB, 0, stream>>>(gates, c_f, h_f, h_b);

        // glimpse attention
        k_gemm_wmma<0, 0><<<g_hh, TB, 0, stream>>>(h_b, WqgT, bq_g, (void*)qg, Bn, Hn, Hn);
        k_update_mask<<<(Bn + TB - 1) / TB, TB, 0, stream>>>(prev, mask);
        k_attn_scores<<<g_attn, TB, 0, stream>>>(qg, e_g, v_g, ug);
        k_masked_softmax<<<Bn / 8, TB, 0, stream>>>(ug, mask, pg);
        k_glimpse<<<g_bh, TB, 0, stream>>>(pg, e_g, gv_f, gv_b);

        // pointer head
        k_gemm_wmma<0, 0><<<g_hh, TB, 0, stream>>>(gv_b, WqpT, bq_p, (void*)qp, Bn, Hn, Hn);
        k_attn_scores<<<g_attn, TB, 0, stream>>>(qp, e_p, v_p, up);
        k_pointer_finalize<<<Bn, TB, 0, stream>>>(up, mask, out_probs, out_sel, prev, t);

        // next decoder input
        k_gather_x<<<(Bn * En + TB - 1) / TB, TB, 0, stream>>>(prev, emb, x_b);
    }

    k_final_copy<<<g_bh, TB, 0, stream>>>(h_f, c_f, out_hy, out_cy);
}
